// topicAttention_38139309588734
// MI455X (gfx1250) — matrix operand order repaired and checked by reading, not compiled, not run on hardware
//
#include <hip/hip_runtime.h>
#include <hip/hip_bf16.h>
#include <math.h>

typedef __attribute__((ext_vector_type(16))) __bf16 v16bf;
typedef __attribute__((ext_vector_type(8)))  float  v8f;

struct bfrag32 { uint4 lo, hi; };   // 32 bytes -> bit_cast to v16bf

__device__ __forceinline__ unsigned short f2bf(float x) {
  union { float f; unsigned u; } c; c.f = x;
  unsigned r = c.u + 0x7FFFu + ((c.u >> 16) & 1u);   // round-to-nearest-even
  return (unsigned short)(r >> 16);
}

// ---------------------------------------------------------------------------
// Kernel 1: theta_w[b,e] = sum_t theta[b,t] * w[e,t]     (B=64,T=100,E=256)
// ---------------------------------------------------------------------------
__global__ __launch_bounds__(256)
void k_theta_w(const float* __restrict__ theta, const float* __restrict__ w,
               float* __restrict__ theta_w) {
  __shared__ float th[128];
  const int b = blockIdx.x, e = threadIdx.x;
  if (e < 100) th[e] = theta[b * 100 + e];
  __syncthreads();
  const float* wr = w + e * 100;
  float acc = 0.f;
#pragma unroll 4
  for (int t = 0; t < 100; ++t) acc += th[t] * wr[t];
  theta_w[b * 256 + e] = acc;
}

// ---------------------------------------------------------------------------
// Kernel 2: per (b, 128-row S tile):
//   pre = h_tile @ u + theta_w[b,:]  (bf16 WMMA, f32 accum)
//   g[b,s] = sum_n tanh(pre[s,n]) * v[n]
// Block: 256 threads (8 waves). Wave tiling: 4 M-strips x 4 N-tiles each
// (sg = wid>>2 selects strip quad, ng = wid&3 selects N-tile quad), so each
// A/B fragment is reused by 4 WMMAs and only 16 ds_load_b128 feed 16 WMMAs
// per K-step.
// LDS: u transposed bf16 [256][264], h tile bf16 [128][264], tw/v/gpart f32.
// ---------------------------------------------------------------------------
#define US 264
#define HS 264

__global__ __launch_bounds__(256)
void k_main(const float* __restrict__ h, const float* __restrict__ u,
            const float* __restrict__ v, const float* __restrict__ theta_w,
            float* __restrict__ g_out) {
  extern __shared__ char smem[];
  unsigned short* uT = (unsigned short*)smem;            // [n][k] : 256*US
  unsigned short* hA = uT + 256 * US;                    // [m][k] : 128*HS
  float* tw = (float*)(hA + 128 * HS);                   // 256
  float* vl = tw + 256;                                  // 256
  float* gp = vl + 256;                                  // 128 partial g rows

  const int tid = threadIdx.x;
  const int b   = blockIdx.x >> 4;          // 16 tiles of 128 rows per batch
  const int s0  = (blockIdx.x & 15) * 128;

  // ---- stage u (transposed) as bf16 ----
  const float4* u4 = (const float4*)u;
#pragma unroll 4
  for (int i = 0; i < 64; ++i) {
    int q  = tid + i * 256;                 // 16384 float4 total
    int k  = q >> 6;                        // contraction index (row of u)
    int f  = (q & 63) * 4;                  // output column
    float4 val = u4[q];
    uT[(f + 0) * US + k] = f2bf(val.x);
    uT[(f + 1) * US + k] = f2bf(val.y);
    uT[(f + 2) * US + k] = f2bf(val.z);
    uT[(f + 3) * US + k] = f2bf(val.w);
  }
  // ---- stage h tile as bf16 (row-major) ----
  const float4* h4 = (const float4*)(h + ((size_t)b * 2048 + s0) * 256);
#pragma unroll 4
  for (int i = 0; i < 32; ++i) {
    int q = tid + i * 256;                  // 8192 float4 total
    int m = q >> 6;
    int e = (q & 63) * 4;
    float4 val = h4[q];
    unsigned short t4[4] = { f2bf(val.x), f2bf(val.y), f2bf(val.z), f2bf(val.w) };
    __builtin_memcpy(&hA[m * HS + e], t4, 8);   // 8B-aligned ds_store_b64
  }
  tw[tid] = theta_w[b * 256 + tid];
  vl[tid] = v[tid];
  if (tid < 128) gp[tid] = 0.f;
  __syncthreads();

  // ---- WMMA GEMM: 4x4 tile grid per wave ----
  const int lane = tid & 31;
  const int wid  = tid >> 5;      // 0..7
  const int gsel = lane >> 4;     // K-half select
  const int l16  = lane & 15;
  const int sg   = wid >> 2;      // strip group: rows sg*64 .. sg*64+63
  const int ng   = wid & 3;       // N-tile group: cols ng*64 .. ng*64+63

  v8f acc[16] = {};               // acc[i*4+j] : strip i, ntile j
  const unsigned short* arow0 = hA + (sg * 64 + l16) * HS;   // + i*16*HS
  const unsigned short* bcol0 = uT + (ng * 64 + l16) * US;   // + j*16*US

#pragma unroll
  for (int k0 = 0; k0 < 256; k0 += 32) {
    // A fragments: 16x32 bf16; lanes 0-15 K-chunks {0..7,16..23}, lanes
    // 16-31 offset +8. Two contiguous 16B ds_load_b128 each.
    v16bf a[4], bm[4];
#pragma unroll
    for (int i = 0; i < 4; ++i) {
      bfrag32 af;
      af.lo = *(const uint4*)(arow0 + i * 16 * HS + k0 + 8 * gsel);
      af.hi = *(const uint4*)(arow0 + i * 16 * HS + k0 + 16 + 8 * gsel);
      a[i] = __builtin_bit_cast(v16bf, af);
    }
#pragma unroll
    for (int j = 0; j < 4; ++j) {
      // B fragment: 32x16 bf16, column n = (ng*4+j)*16 + l16, K chunks {8*gsel, 16+8*gsel}
      const unsigned short* bp = bcol0 + j * 16 * US + k0 + 8 * gsel;
      bfrag32 bf;
      bf.lo = *(const uint4*)(bp);
      bf.hi = *(const uint4*)(bp + 16);
      bm[j] = __builtin_bit_cast(v16bf, bf);
    }
#pragma unroll
    for (int i = 0; i < 4; ++i)
#pragma unroll
      for (int j = 0; j < 4; ++j)
        acc[i * 4 + j] = __builtin_amdgcn_wmma_f32_16x16x32_bf16(
            false, a[i], false, bm[j], (short)0, acc[i * 4 + j], false, false);
  }

  // ---- fused epilogue: g contribution = tanh(acc + tw[n]) * v[n] ----
  // C layout: lane l16 = N within tile; VGPR r -> M = 8*gsel + r.
#pragma unroll
  for (int i = 0; i < 4; ++i) {
    float rsum[8] = {0.f,0.f,0.f,0.f,0.f,0.f,0.f,0.f};
#pragma unroll
    for (int j = 0; j < 4; ++j) {
      int n = (ng * 4 + j) * 16 + l16;
      float twv = tw[n];
      float vv  = vl[n];
#pragma unroll
      for (int r = 0; r < 8; ++r)
        rsum[r] += tanhf(acc[i * 4 + j][r] + twv) * vv;
    }
#pragma unroll
    for (int r = 0; r < 8; ++r) {
      float s = rsum[r];
      s += __shfl_xor(s, 1, 32);
      s += __shfl_xor(s, 2, 32);
      s += __shfl_xor(s, 4, 32);
      s += __shfl_xor(s, 8, 32);   // full sum over the 16-lane half-wave
      if (l16 == 0)                // 4 ng-groups accumulate into each row
        atomicAdd(&gp[sg * 64 + i * 16 + 8 * gsel + r], s);
    }
  }
  __syncthreads();
  if (tid < 128) g_out[(size_t)b * 2048 + s0 + tid] = gp[tid];
}

// ---------------------------------------------------------------------------
// Kernel 3: in-place softmax over S=2048 per batch row
// ---------------------------------------------------------------------------
__global__ __launch_bounds__(256)
void k_softmax(float* __restrict__ g) {
  __shared__ float red[256];
  const int b = blockIdx.x, t = threadIdx.x;
  float* row = g + (size_t)b * 2048;
  float vals[8];
  float m = -3.402823466e38f;
#pragma unroll
  for (int i = 0; i < 8; ++i) { vals[i] = row[t + i * 256]; m = fmaxf(m, vals[i]); }
  red[t] = m; __syncthreads();
  for (int s = 128; s > 0; s >>= 1) {
    if (t < s) red[t] = fmaxf(red[t], red[t + s]);
    __syncthreads();
  }
  m = red[0];
  __syncthreads();
  float sum = 0.f;
#pragma unroll
  for (int i = 0; i < 8; ++i) { vals[i] = __expf(vals[i] - m); sum += vals[i]; }
  red[t] = sum; __syncthreads();
  for (int s = 128; s > 0; s >>= 1) {
    if (t < s) red[t] += red[t + s];
    __syncthreads();
  }
  float inv = 1.0f / red[0];
#pragma unroll
  for (int i = 0; i < 8; ++i) row[t + i * 256] = vals[i] * inv;
}

// ---------------------------------------------------------------------------
// Kernel 4: vec[b,e] = sum_s weight[b,s] * h[b,s,e]   (bandwidth bound)
// ---------------------------------------------------------------------------
__global__ __launch_bounds__(256)
void k_wsum(const float* __restrict__ h, const float* __restrict__ wgt,
            float* __restrict__ out) {
  __shared__ float wl[2048];
  const int b = blockIdx.x, e = threadIdx.x;
#pragma unroll
  for (int i = 0; i < 8; ++i) wl[e + i * 256] = wgt[(size_t)b * 2048 + e + i * 256];
  __syncthreads();
  const float* hb = h + (size_t)b * 2048 * 256 + e;
  float a0 = 0.f, a1 = 0.f, a2 = 0.f, a3 = 0.f;
  for (int s = 0; s < 2048; s += 4) {
    __builtin_prefetch(hb + (size_t)(s + 16) * 256, 0, 1);  // global_prefetch
    a0 += wl[s + 0] * hb[(size_t)(s + 0) * 256];
    a1 += wl[s + 1] * hb[(size_t)(s + 1) * 256];
    a2 += wl[s + 2] * hb[(size_t)(s + 2) * 256];
    a3 += wl[s + 3] * hb[(size_t)(s + 3) * 256];
  }
  out[b * 256 + e] = (a0 + a1) + (a2 + a3);
}

// ---------------------------------------------------------------------------
extern "C" void kernel_launch(void* const* d_in, const int* in_sizes, int n_in,
                              void* d_out, int out_size, void* d_ws, size_t ws_size,
                              hipStream_t stream) {
  const float* h     = (const float*)d_in[0];   // [64,2048,256]
  const float* theta = (const float*)d_in[1];   // [64,100]
  const float* w     = (const float*)d_in[2];   // [256,100]
  const float* v     = (const float*)d_in[3];   // [256,1]
  const float* u     = (const float*)d_in[4];   // [256,256]
  float* out = (float*)d_out;                   // [64,256]

  float* tw = (float*)d_ws;                     // 64*256 f32
  float* g  = tw + 64 * 256;                    // 64*2048 f32 (then weights in place)

  const size_t smem_main =
      (size_t)(256 * US + 128 * HS) * sizeof(unsigned short) +
      (256 + 256 + 128) * sizeof(float);
  hipFuncSetAttribute((const void*)k_main,
                      hipFuncAttributeMaxDynamicSharedMemorySize, (int)smem_main);

  k_theta_w<<<64, 256, 0, stream>>>(theta, w, tw);
  k_main<<<64 * 16, 256, smem_main, stream>>>(h, u, v, tw, g);
  k_softmax<<<64, 256, 0, stream>>>(g);
  k_wsum<<<64, 256, 0, stream>>>(h, g, out);
}
